// SelectiveAttentionModule_67456756351555
// MI455X (gfx1250) — matrix operand order repaired and checked by reading, not compiled, not run on hardware
//
#include <hip/hip_runtime.h>
#include <hip/hip_bf16.h>

#define BB 8
#define SS 2048
#define EE 768
#define DD 128

typedef _Float16 v16h __attribute__((ext_vector_type(16)));
typedef _Float16 v8h  __attribute__((ext_vector_type(8)));
typedef float    v8f  __attribute__((ext_vector_type(8)));

// ---------------------------------------------------------------- conversion
__global__ void __launch_bounds__(256)
sam_cvt_f32_f16(const float* __restrict__ src, _Float16* __restrict__ dst, int n) {
    int i = blockIdx.x * blockDim.x + threadIdx.x;
    int stride = gridDim.x * blockDim.x;
    for (; i < n; i += stride) dst[i] = (_Float16)src[i];
}

// ---------------------------------------------------------------- QKV GEMM
// out[m, n] = sum_e A[m,e] * W[n,e] + bias[n]
// one wave per 16x16 output tile; K-steps of 32 via v_wmma_f32_16x16x32_f16
__global__ void __launch_bounds__(256)
sam_qkv_gemm(const _Float16* __restrict__ A,   // [BB*SS, EE] f16
             const _Float16* __restrict__ W,   // [DD, EE] f16
             const float* __restrict__ bias,   // [DD]
             float* __restrict__ out)          // [BB*SS, DD] f32
{
    const int lane = threadIdx.x & 31;
    const int l15  = lane & 15;
    const int hi   = lane >> 4;
    const int wave = blockIdx.x * (blockDim.x >> 5) + (threadIdx.x >> 5);
    const int tn = wave & 7;     // DD/16 = 8 column tiles
    const int tm = wave >> 3;    // (BB*SS)/16 = 1024 row tiles
    const int m0 = tm * 16, n0 = tn * 16;

    const _Float16* arow = A + (size_t)(m0 + l15) * EE;
    const _Float16* wrow = W + (size_t)(n0 + l15) * EE;

    v8f c = {};
#pragma unroll 4
    for (int e0 = 0; e0 < EE; e0 += 32) {
        // A fragment: lane = M row; lanes 0-15 own K 0..15, 16-31 own K 16..31 (contiguous)
        v8h alo = *(const v8h*)(arow + e0 + hi * 16);
        v8h ahi = *(const v8h*)(arow + e0 + hi * 16 + 8);
        v16h a;
#pragma unroll
        for (int i = 0; i < 8; i++) { a[i] = alo[i]; a[i + 8] = ahi[i]; }
        // B fragment: lane = N column; lanes 0-15 K=0..15, lanes 16-31 K=16..31 (contiguous)
        v16h b = *(const v16h*)(wrow + e0 + hi * 16);
        c = __builtin_amdgcn_wmma_f32_16x16x32_f16(false, a, false, b, (short)0, c, false, false);
    }
    const float bv = bias[n0 + l15];
#pragma unroll
    for (int i = 0; i < 8; i++) {
        out[(size_t)(m0 + i + 8 * hi) * DD + n0 + l15] = c[i] + bv;
    }
}

// ---------------------------------------------------------------- gating
__device__ __forceinline__ float sam_wave_sum32(float v) {
#pragma unroll
    for (int off = 1; off < 32; off <<= 1) v += __shfl_xor(v, off, 32);
    return v;
}

__global__ void __launch_bounds__(256)
sam_gate(const float* __restrict__ Qf, const float* __restrict__ Kf, const float* __restrict__ Vf,
         const float* __restrict__ alpha, const float* __restrict__ tokenq, const float* __restrict__ tokenv,
         _Float16* __restrict__ Qg,      // [BB*SS, DD] gated Q * 1/sqrt(dk)
         _Float16* __restrict__ Kh,      // [BB*SS, DD]
         _Float16* __restrict__ VgT)     // [BB, DD, SS] gated V, transposed
{
    const int lane = threadIdx.x & 31;
    const int row  = blockIdx.x * (blockDim.x >> 5) + (threadIdx.x >> 5);
    const int b = row >> 11;          // SS = 2048
    const int s = row & (SS - 1);

    const float* q = Qf + (size_t)row * DD;
    const float* k = Kf + (size_t)row * DD;
    const float* v = Vf + (size_t)row * DD;

    float dq = 0.f, dv = 0.f;
    float qv[4], vv[4], kv[4];
#pragma unroll
    for (int j = 0; j < 4; j++) {
        const int d = lane * 4 + j;
        float x = q[d]; qv[j] = x;
        float gx = 0.5f * x * (1.f + erff(x * 0.70710678118654752f));   // exact GELU
        dq += tokenq[(size_t)s * DD + d] * gx;
        float y = v[d]; vv[j] = y;
        float gy = 0.5f * y * (1.f + erff(y * 0.70710678118654752f));
        dv += tokenv[(size_t)s * DD + d] * gy;
        kv[j] = k[d];
    }
    dq = sam_wave_sum32(dq);
    dv = sam_wave_sum32(dv);

    const float a  = alpha[s];
    const float pt = 1.f + (1.f / (1.f + expf(-a))) * logf((float)(s + 1));
    const float tq = pt + tanhf(dq);
    const float tv = pt + tanhf(dv);
    const float qscale = 0.08838834764831845f;   // 1/sqrt(128)

#pragma unroll
    for (int j = 0; j < 4; j++) {
        const int d = lane * 4 + j;
        Qg[(size_t)row * DD + d] = (_Float16)(tq * qv[j] * qscale);
        Kh[(size_t)row * DD + d] = (_Float16)kv[j];
        VgT[((size_t)b * DD + d) * SS + s] = (_Float16)(tv * vv[j]);
    }
}

// ---------------------------------------------------------------- attention
// One wave per (batch, 16-row query tile). Flash-style online softmax.
// Key tiles processed in PAIRS so the PV WMMA runs at full K=32.
__global__ void __launch_bounds__(256)
sam_attn(const _Float16* __restrict__ Qg, const _Float16* __restrict__ Kh,
         const _Float16* __restrict__ VgT, float* __restrict__ out)
{
    __shared__ _Float16 qtile[8][16 * 128];  // 4KB/wave: async-staged Q tile
    __shared__ _Float16 ptile[8][16][32];    // 1KB/wave: P (two key tiles wide)

    const int lane = threadIdx.x & 31;
    const int l15  = lane & 15;
    const int hi   = lane >> 4;
    const int w    = threadIdx.x >> 5;
    const int wave = blockIdx.x * 8 + w;
    const int b  = wave >> 7;              // SS/16 = 128 q-tiles per batch
    const int qt = wave & 127;
    const int q0 = qt * 16;

    // ---- async-stage the contiguous 16x128 (4KB) Q tile into LDS ----------
    {
        const _Float16* qsrc = Qg + ((size_t)b * SS + q0) * DD;
        unsigned lbase = (unsigned)(size_t)(&qtile[w][0]);
#pragma unroll
        for (int t = 0; t < 8; t++) {
            unsigned loff = lbase + (unsigned)(t * 512 + lane * 16);
            unsigned long long gaddr =
                (unsigned long long)(size_t)qsrc + (unsigned)(t * 512 + lane * 16);
            asm volatile("global_load_async_to_lds_b128 %0, %1, off"
                         :: "v"(loff), "v"(gaddr) : "memory");
        }
        asm volatile("s_wait_asynccnt 0" ::: "memory");
    }

    // persistent Q A-fragments: 4 K-steps of 32 covering d=0..127 (from LDS)
    v16h aq[4];
    {
        const _Float16* qrow = &qtile[w][l15 * 128];
#pragma unroll
        for (int t = 0; t < 4; t++) {
            v8h lo = *(const v8h*)(qrow + t * 32 + hi * 16);
            v8h hh = *(const v8h*)(qrow + t * 32 + hi * 16 + 8);
#pragma unroll
            for (int i = 0; i < 8; i++) { aq[t][i] = lo[i]; aq[t][i + 8] = hh[i]; }
        }
    }

    v8f vzero = {};
    v8f acc[8];
#pragma unroll
    for (int j = 0; j < 8; j++) acc[j] = vzero;
    float rm[8], rl[8];
#pragma unroll
    for (int i = 0; i < 8; i++) { rm[i] = -1e30f; rl[i] = 0.f; }

    for (int kt = 0; kt <= qt; kt += 2) {
        const bool has1 = (kt + 1 <= qt);
        const _Float16* krow0 = Kh + ((size_t)b * SS + kt * 16 + l15) * DD;
        const _Float16* krow1 = krow0 + 16 * DD;
        if (kt + 2 <= qt)
            __builtin_prefetch(krow0 + 32 * DD, 0, 3);

        // score tiles: S0 = Q.K0^T, S1 = Q.K1^T  (each 16x16, K-dim 128)
        v8f sc0 = vzero, sc1 = vzero;
#pragma unroll
        for (int t = 0; t < 4; t++) {
            v16h bk = *(const v16h*)(krow0 + t * 32 + hi * 16);
            sc0 = __builtin_amdgcn_wmma_f32_16x16x32_f16(false, aq[t], false, bk, (short)0, sc0, false, false);
        }
        if (has1) {
#pragma unroll
            for (int t = 0; t < 4; t++) {
                v16h bk = *(const v16h*)(krow1 + t * 32 + hi * 16);
                sc1 = __builtin_amdgcn_wmma_f32_16x16x32_f16(false, aq[t], false, bk, (short)0, sc1, false, false);
            }
        }

        // causal masking: element (m = i + 8*hi, n = l15) of each tile
        if (kt == qt) {          // tile0 is the diagonal tile (tile1 absent)
#pragma unroll
            for (int i = 0; i < 8; i++)
                if (l15 > (i + 8 * hi)) sc0[i] = -1e30f;
        }
        if (!has1) {
#pragma unroll
            for (int i = 0; i < 8; i++) sc1[i] = -1e30f;   // exp -> exactly 0
        } else if (kt + 1 == qt) {   // tile1 is the diagonal tile
#pragma unroll
            for (int i = 0; i < 8; i++)
                if (l15 > (i + 8 * hi)) sc1[i] = -1e30f;
        }

        // online softmax over the 32 columns of this pair
        float corr[8];
#pragma unroll
        for (int i = 0; i < 8; i++) {
            float mv = fmaxf(sc0[i], sc1[i]);
#pragma unroll
            for (int off = 1; off < 16; off <<= 1) mv = fmaxf(mv, __shfl_xor(mv, off, 32));
            const float mn = fmaxf(rm[i], mv);
            corr[i] = expf(rm[i] - mn);
            rm[i] = mn;
            const float p0 = expf(sc0[i] - mn);
            const float p1 = expf(sc1[i] - mn);
            sc0[i] = p0; sc1[i] = p1;
            float su = p0 + p1;
#pragma unroll
            for (int off = 1; off < 16; off <<= 1) su += __shfl_xor(su, off, 32);
            rl[i] = rl[i] * corr[i] + su;
        }
#pragma unroll
        for (int j = 0; j < 8; j++) {
#pragma unroll
            for (int i = 0; i < 8; i++) acc[j][i] *= corr[i];
        }

        // bounce 16x32 P through LDS: C-layout -> A-layout
#pragma unroll
        for (int i = 0; i < 8; i++) {
            ptile[w][i + 8 * hi][l15]      = (_Float16)sc0[i];
            ptile[w][i + 8 * hi][16 + l15] = (_Float16)sc1[i];
        }
        asm volatile("s_wait_dscnt 0" ::: "memory");

        v16h pa;
        {
            v8h lo = *(const v8h*)(&ptile[w][l15][hi * 16]);
            v8h hh = *(const v8h*)(&ptile[w][l15][hi * 16 + 8]);
#pragma unroll
            for (int i = 0; i < 8; i++) { pa[i] = lo[i]; pa[i + 8] = hh[i]; }
        }

        // O += P.Vg with full K=32: lanes 0-15 carry keys kt*16..+15,
        // lanes 16-31 carry keys kt*16+16..+31 (contribution 0 if tile1 masked)
#pragma unroll
        for (int j = 0; j < 8; j++) {
            v16h bv = *(const v16h*)(VgT + ((size_t)b * DD + j * 16 + l15) * SS + kt * 16 + hi * 16);
            acc[j] = __builtin_amdgcn_wmma_f32_16x16x32_f16(false, pa, false, bv, (short)0, acc[j], false, false);
        }
    }

    // normalize and write out
#pragma unroll
    for (int j = 0; j < 8; j++) {
#pragma unroll
        for (int i = 0; i < 8; i++) {
            out[((size_t)b * SS + q0 + i + 8 * hi) * DD + j * 16 + l15] = acc[j][i] / rl[i];
        }
    }
}

// ---------------------------------------------------------------- launch
extern "C" void kernel_launch(void* const* d_in, const int* in_sizes, int n_in,
                              void* d_out, int out_size, void* d_ws, size_t ws_size,
                              hipStream_t stream) {
    const float* x      = (const float*)d_in[0];
    const float* Wq_w   = (const float*)d_in[1];
    const float* Wq_b   = (const float*)d_in[2];
    const float* Wk_w   = (const float*)d_in[3];
    const float* Wk_b   = (const float*)d_in[4];
    const float* Wv_w   = (const float*)d_in[5];
    const float* Wv_b   = (const float*)d_in[6];
    const float* alpha  = (const float*)d_in[7];
    const float* tokenq = (const float*)d_in[8];
    const float* tokenv = (const float*)d_in[9];
    float* out = (float*)d_out;

    char* ws = (char*)d_ws;
    size_t off = 0;
    _Float16* xh  = (_Float16*)(ws + off); off += (size_t)BB * SS * EE * 2;   // 25,165,824
    _Float16* wqh = (_Float16*)(ws + off); off += (size_t)DD * EE * 2;
    _Float16* wkh = (_Float16*)(ws + off); off += (size_t)DD * EE * 2;
    _Float16* wvh = (_Float16*)(ws + off); off += (size_t)DD * EE * 2;
    float* Qf = (float*)(ws + off); off += (size_t)BB * SS * DD * 4;
    float* Kf = (float*)(ws + off); off += (size_t)BB * SS * DD * 4;
    float* Vf = (float*)(ws + off); off += (size_t)BB * SS * DD * 4;
    _Float16* Qg  = (_Float16*)(ws + off); off += (size_t)BB * SS * DD * 2;
    _Float16* Khp = (_Float16*)(ws + off); off += (size_t)BB * SS * DD * 2;
    _Float16* VgT = (_Float16*)(ws + off); off += (size_t)BB * SS * DD * 2;
    (void)ws_size; (void)in_sizes; (void)n_in; (void)out_size;

    // 1) f32 -> f16 conversions
    sam_cvt_f32_f16<<<4096, 256, 0, stream>>>(x, xh, BB * SS * EE);
    sam_cvt_f32_f16<<<384, 256, 0, stream>>>(Wq_w, wqh, DD * EE);
    sam_cvt_f32_f16<<<384, 256, 0, stream>>>(Wk_w, wkh, DD * EE);
    sam_cvt_f32_f16<<<384, 256, 0, stream>>>(Wv_w, wvh, DD * EE);

    // 2) QKV projections: (BB*SS/16)*(DD/16) = 8192 tiles, 8 waves/block
    sam_qkv_gemm<<<1024, 256, 0, stream>>>(xh, wqh, Wq_b, Qf);
    sam_qkv_gemm<<<1024, 256, 0, stream>>>(xh, wkh, Wk_b, Kf);
    sam_qkv_gemm<<<1024, 256, 0, stream>>>(xh, wvh, Wv_b, Vf);

    // 3) temperatures + gating, V stored transposed
    sam_gate<<<2048, 256, 0, stream>>>(Qf, Kf, Vf, alpha, tokenq, tokenv, Qg, Khp, VgT);

    // 4) causal flash attention: 8 batches * 128 q-tiles = 1024 waves, 8/block
    sam_attn<<<128, 256, 0, stream>>>(Qg, Khp, VgT, out);
}
